// LinearMultiHeadAttention_17171279249469
// MI455X (gfx1250) — compile-verified, matrix operand order checked by reading, not run on hardware
//
#include <hip/hip_runtime.h>

typedef _Float16 half_t;
typedef __attribute__((ext_vector_type(4)))  _Float16 v4h;
typedef __attribute__((ext_vector_type(8)))  _Float16 v8h;
typedef __attribute__((ext_vector_type(16))) _Float16 v16h;
typedef __attribute__((ext_vector_type(8)))  float    v8f;
typedef __attribute__((ext_vector_type(4)))  float    v4f;
typedef __attribute__((ext_vector_type(4)))  unsigned int u32x4;
typedef __attribute__((ext_vector_type(8)))  int      i32x8;
typedef __attribute__((ext_vector_type(4)))  int      i32x4;

#define B_  4
#define S_  4096
#define D_  1024
#define H_  16
#define HD_ 64
#define EPS_ 1e-6f

// ---------------------------------------------------------------------------
// WMMA fragment load (wave32, v_wmma_f32_16x16x32_f16).
// ISA 7.12.2, 16-bit A (16x32 MxK): lanes 0-15 -> M=lane, K = {0..7,16..23};
// lanes 16-31 -> M=lane-16, K = {8..15,24..31}.  B (32x16 KxN) symmetric with
// lane = N.  Per lane: two contiguous 16-byte runs -> 2x (ds|global)_load_b128.
// ---------------------------------------------------------------------------
__device__ __forceinline__ v16h frag_ld(const half_t* p, int ld) {
  const int lane = threadIdx.x & 31;
  const int m  = lane & 15;
  const int kb = (lane >> 4) << 3;   // 0 or 8
  const half_t* r = p + m * ld + kb;
  v8h lo = *(const v8h*)(r);         // K = kb + 0..7
  v8h hi = *(const v8h*)(r + 16);    // K = kb + 16..23
  return __builtin_shufflevector(lo, hi, 0, 1, 2, 3, 4, 5, 6, 7,
                                 8, 9, 10, 11, 12, 13, 14, 15);
}

__device__ __forceinline__ v8f wmma_f16(v16h a, v16h b, v8f c) {
  return __builtin_amdgcn_wmma_f32_16x16x32_f16(
      /*neg_a=*/false, a, /*neg_b=*/false, b,
      /*c_mod=*/(short)0, c, /*reuse_a=*/false, /*reuse_b=*/false);
}

// ---------------------------------------------------------------------------
// TDM: DMA a 2D f16 tile (tile_w x tile_h elems) from a row-major tensor
// (row stride = row_stride elems) straight into LDS, packed row-major.
// D# layout per CDNA5 ISA ch.8: group0 = {count|flags, lds_addr, gaddr_lo,
// gaddr_hi|type=2}; group1 = {data_size=2B, dims/tile/stride}; 2D tile ->
// groups 2/3 zero.  Tracked by TENSORcnt.  Issue from ONE wave per block.
// ---------------------------------------------------------------------------
__device__ __forceinline__ void tdm_load_2d(const void* lds_dst, const void* gsrc,
                                            unsigned tile_w, unsigned tile_h,
                                            unsigned tensor_w, unsigned tensor_h,
                                            unsigned row_stride) {
  unsigned long long ga = (unsigned long long)(size_t)gsrc;
  unsigned lds_addr = (unsigned)(size_t)lds_dst;   // LDS aperture: low 32 bits
  u32x4 g0;
  g0[0] = 1u;                                        // count=1 (valid user D#)
  g0[1] = lds_addr;                                  // lds_addr (bytes)
  g0[2] = (unsigned)(ga & 0xffffffffu);              // global_addr[31:0]
  g0[3] = (unsigned)((ga >> 32) & 0x1ffffffu) | (2u << 30);  // [56:32] | type=2
  i32x8 g1;
  g1[0] = 0x00010000;                                // data_size=1 -> 2 bytes
  g1[1] = (int)((tensor_w & 0xffffu) << 16);         // tensor_dim0[15:0]
  g1[2] = (int)(((tensor_w >> 16) & 0xffffu) | ((tensor_h & 0xffffu) << 16));
  g1[3] = (int)(((tensor_h >> 16) & 0xffffu) | (tile_w << 16));  // tile_dim0
  g1[4] = (int)tile_h;                               // tile_dim1, tile_dim2=0
  g1[5] = (int)row_stride;                           // tensor_dim0_stride lo32
  g1[6] = 0;                                         // stride hi16, dim1_stride
  g1[7] = 0;
  i32x4 z4 = {0, 0, 0, 0};
#if __clang_major__ >= 23
  i32x8 z8 = {0, 0, 0, 0, 0, 0, 0, 0};
  __builtin_amdgcn_tensor_load_to_lds(g0, g1, z4, z4, z8, 0);
#else
  __builtin_amdgcn_tensor_load_to_lds(g0, g1, z4, z4, 0);
#endif
}

// ---------------------------------------------------------------------------
// One-pass f32 -> f16 conversion (weights), 8 elems/thread, b128 in/out.
// ---------------------------------------------------------------------------
__global__ __launch_bounds__(256) void cvt_f32_f16(const float* __restrict__ x,
                                                   half_t* __restrict__ y, int n) {
  int i = (blockIdx.x * 256 + threadIdx.x) * 8;
  if (i < n) {
    v4f a = *(const v4f*)(x + i);
    v4f b = *(const v4f*)(x + i + 4);
    v8h o;
#pragma unroll
    for (int j = 0; j < 4; ++j) { o[j] = (half_t)a[j]; o[4 + j] = (half_t)b[j]; }
    *(v8h*)(y + i) = o;
  }
}

// ---------------------------------------------------------------------------
// Y = act(X @ W16^T + bias).  X:[M,K] f32 or f16, W16:[N,K] f16 row-major.
// Block tile 128x128, BK=32, 8 waves, wave tile 32(M) x 64(N).
// Double-buffered LDS; W tile (and X tile when f16) DMA'd by the TDM and
// overlapped with WMMA on the previous tile (s_wait_tensorcnt pipelining).
// ELU: y = elu(x)+1 = x>0 ? x+1 : exp(x).  OUT32: write f32 else f16.
// ---------------------------------------------------------------------------
template <typename XT, bool ELU, bool OUT32>
__global__ __launch_bounds__(256) void proj_gemm(
    const XT* __restrict__ X, const half_t* __restrict__ W16,
    const float* __restrict__ bias, void* __restrict__ Yv,
    int M, int N, int K) {
  __shared__ half_t As[2][128 * 32];
  __shared__ half_t Bs[2][128 * 32];

  constexpr bool XF16 = (sizeof(XT) == 2);
  constexpr int  NOPS = XF16 ? 2 : 1;   // TDM ops issued per stage

  const int tid  = threadIdx.x;
  const int lane = tid & 31;
  const int wid  = tid >> 5;
  const int wm   = wid >> 1;            // 0..3 -> M offset wm*32
  const int wn   = wid & 1;             // 0..1 -> N offset wn*64
  const int m0   = blockIdx.y * 128;
  const int n0   = blockIdx.x * 128;
  const int NIT  = K / 32;

  // ---- stage 0 ----
  if (wid == 0) {
    tdm_load_2d(&Bs[0][0], W16 + (size_t)n0 * K, 32, 128, K, N, K);
    if (XF16) tdm_load_2d(&As[0][0], (const half_t*)X + (size_t)m0 * K, 32, 128, K, M, K);
  }
  if constexpr (!XF16) {
#pragma unroll
    for (int i = 0; i < 4; ++i) {
      int qi = tid + 256 * i, row = qi >> 3, c4 = (qi & 7) * 4;
      const XT* xa = X + (size_t)(m0 + row) * K + c4;
      v4h av;
#pragma unroll
      for (int j = 0; j < 4; ++j) av[j] = (half_t)(float)xa[j];
      *(v4h*)&As[0][row * 32 + c4] = av;
    }
  }

  v8f acc[2][4] = {};

  for (int it = 0; it < NIT; ++it) {
    const int cur = it & 1;
    const int kk  = it * 32;
    // ---- issue / stage next tile while current computes ----
    if (it + 1 < NIT) {
      if (wid == 0) {
        tdm_load_2d(&Bs[1 - cur][0], W16 + (size_t)n0 * K + kk + 32, 32, 128, K, N, K);
        if (XF16)
          tdm_load_2d(&As[1 - cur][0], (const half_t*)X + (size_t)m0 * K + kk + 32,
                      32, 128, K, M, K);
      }
      if constexpr (!XF16) {
#pragma unroll
        for (int i = 0; i < 4; ++i) {
          int qi = tid + 256 * i, row = qi >> 3, c4 = (qi & 7) * 4;
          const XT* xa = X + (size_t)(m0 + row) * K + kk + 32 + c4;
          __builtin_prefetch(xa + 32, 0, 3);   // global_prefetch_b8
          v4h av;
#pragma unroll
          for (int j = 0; j < 4; ++j) av[j] = (half_t)(float)xa[j];
          *(v4h*)&As[1 - cur][row * 32 + c4] = av;
        }
      }
    }
    // current tile's DMA done; next tile's NOPS DMAs stay in flight
    if (wid == 0) {
      if (it + 1 < NIT) __builtin_amdgcn_s_wait_tensorcnt(NOPS);
      else              __builtin_amdgcn_s_wait_tensorcnt(0);
    }
    __syncthreads();

    v16h a[2], b[4];
#pragma unroll
    for (int mi = 0; mi < 2; ++mi)
      a[mi] = frag_ld(&As[cur][(wm * 32 + mi * 16) * 32], 32);
#pragma unroll
    for (int ni = 0; ni < 4; ++ni)
      b[ni] = frag_ld(&Bs[cur][(wn * 64 + ni * 16) * 32], 32);
#pragma unroll
    for (int mi = 0; mi < 2; ++mi)
#pragma unroll
      for (int ni = 0; ni < 4; ++ni)
        acc[mi][ni] = wmma_f16(a[mi], b[ni], acc[mi][ni]);
    __syncthreads();   // protect buffer reuse by next stage
  }

  const int hi8 = (lane >> 4) * 8;      // C/D layout: lanes 16-31 hold M+8
#pragma unroll
  for (int mi = 0; mi < 2; ++mi)
#pragma unroll
    for (int ni = 0; ni < 4; ++ni) {
      int col = n0 + wn * 64 + ni * 16 + (lane & 15);
      float bvv = bias[col];
#pragma unroll
      for (int r = 0; r < 8; ++r) {
        int row = m0 + wm * 32 + mi * 16 + hi8 + r;
        float v = acc[mi][ni][r] + bvv;
        if (ELU) v = (v > 0.f) ? (v + 1.f) : __expf(v);
        if (OUT32)
          ((float*)Yv)[(size_t)row * N + col] = v;
        else
          ((half_t*)Yv)[(size_t)row * N + col] = (half_t)v;
      }
    }
}

// ---------------------------------------------------------------------------
// Per (b,h): KV[d,e] = sum_s phi_k[s,d] * v[s,e]  (64x64, K=4096)
//            Ksum[d] = sum_s phi_k[s,d]
// k/v tiles staged TRANSPOSED (d-major / e-major) so fragments are b128 reads.
// 8 waves: wave pair shares a 32-row s-chunk, splits the e tiles; cross-wave
// reduction via ds_add_f32.
// ---------------------------------------------------------------------------
__global__ __launch_bounds__(256) void kv_ksum_kernel(
    const half_t* __restrict__ Kf, const half_t* __restrict__ Vf,
    const unsigned char* __restrict__ mask,
    float* __restrict__ KV, float* __restrict__ Ksum) {
  __shared__ half_t KsT[64 * 128];    // [d][s_local]
  __shared__ half_t VsT[64 * 128];    // [e][s_local]
  __shared__ float  KVred[64 * 64];
  __shared__ float  Ksred[64];

  const int tid  = threadIdx.x;
  const int lane = tid & 31;
  const int wid  = tid >> 5;
  const int bh   = blockIdx.x;
  const int b    = bh >> 4;
  const int h    = bh & 15;

  for (int i = tid; i < 4096; i += 256) KVred[i] = 0.f;
  if (tid < 64) Ksred[tid] = 0.f;

  const int d8 = (tid & 7) * 8;
  float ksl[8] = {0.f, 0.f, 0.f, 0.f, 0.f, 0.f, 0.f, 0.f};

  const int soff = (wid >> 1) * 32;
  const int etb  = (wid & 1) * 2;
  v8f acc[4][2] = {};

  for (int sb = 0; sb < S_; sb += 128) {
    __syncthreads();
#pragma unroll
    for (int p = 0; p < 4; ++p) {
      int row = (tid >> 3) + 32 * p;
      int s   = sb + row;
      size_t g = (size_t)(b * S_ + s) * D_ + h * HD_ + d8;
      float mm = mask[b * S_ + s] ? 1.f : 0.f;
      v8h kvec = *(const v8h*)(Kf + g);
      v8h vvec = *(const v8h*)(Vf + g);
#pragma unroll
      for (int j = 0; j < 8; ++j) {
        float kv = (float)kvec[j] * mm;
        ksl[j] += kv;
        KsT[(d8 + j) * 128 + row] = (half_t)kv;
        VsT[(d8 + j) * 128 + row] = vvec[j];
      }
    }
    __syncthreads();

    v16h a[4], bb[2];
#pragma unroll
    for (int dt = 0; dt < 4; ++dt)
      a[dt] = frag_ld(&KsT[(dt * 16) * 128 + soff], 128);
#pragma unroll
    for (int et = 0; et < 2; ++et)
      bb[et] = frag_ld(&VsT[((etb + et) * 16) * 128 + soff], 128);
#pragma unroll
    for (int dt = 0; dt < 4; ++dt)
#pragma unroll
      for (int et = 0; et < 2; ++et)
        acc[dt][et] = wmma_f16(a[dt], bb[et], acc[dt][et]);
  }

  __syncthreads();
  const int hi8 = (lane >> 4) * 8;
#pragma unroll
  for (int dt = 0; dt < 4; ++dt)
#pragma unroll
    for (int et = 0; et < 2; ++et) {
      int e = (etb + et) * 16 + (lane & 15);
#pragma unroll
      for (int r = 0; r < 8; ++r) {
        int d = dt * 16 + hi8 + r;
        atomicAdd(&KVred[d * 64 + e], acc[dt][et][r]);   // ds_add_f32
      }
    }
#pragma unroll
  for (int j = 0; j < 8; ++j) atomicAdd(&Ksred[d8 + j], ksl[j]);
  __syncthreads();

  float* KVo = KV + (size_t)bh * 4096;
  for (int i = tid; i < 4096; i += 256) KVo[i] = KVred[i];
  if (tid < 64) Ksum[bh * 64 + tid] = Ksred[tid];
}

// ---------------------------------------------------------------------------
// V_new[s,e] = Z[s] * sum_d phi_q[s,d] * KV[d,e],  Z = 1/(phi_q . (Ksum+eps))
// ---------------------------------------------------------------------------
__global__ __launch_bounds__(256) void apply_kv_kernel(
    const half_t* __restrict__ Qf, const float* __restrict__ KV,
    const float* __restrict__ Ksum, half_t* __restrict__ Vn) {
  __shared__ half_t KVsT[64 * 64];    // [e][d]
  __shared__ float  Zs[128];

  const int tid   = threadIdx.x;
  const int lane  = tid & 31;
  const int wid   = tid >> 5;
  const int bh    = blockIdx.y;
  const int b     = bh >> 4;
  const int h     = bh & 15;
  const int rows0 = blockIdx.x * 128;

  const float* KVg = KV + (size_t)bh * 4096;
  for (int i = tid; i < 4096; i += 256) {
    int d = i >> 6, e = i & 63;
    KVsT[e * 64 + d] = (half_t)KVg[i];
  }

  if (tid < 128) {
    int s = rows0 + tid;
    const v8h*   q8 = (const v8h*)(Qf + (size_t)(b * S_ + s) * D_ + h * HD_);
    const float* ks = Ksum + bh * 64;
    float dot = 0.f;
#pragma unroll
    for (int i = 0; i < 8; ++i) {
      v8h qq = q8[i];
#pragma unroll
      for (int j = 0; j < 8; ++j) dot += (float)qq[j] * (ks[i * 8 + j] + EPS_);
    }
    Zs[tid] = 1.f / dot;
  }
  __syncthreads();

  v8f acc[4] = {};
  const half_t* qbase = Qf + (size_t)(b * S_ + rows0 + wid * 16) * D_ + h * HD_;
#pragma unroll
  for (int kk = 0; kk < 64; kk += 32) {
    v16h a = frag_ld(qbase + kk, D_);
#pragma unroll
    for (int et = 0; et < 4; ++et) {
      v16h bb = frag_ld(&KVsT[(et * 16) * 64 + kk], 64);
      acc[et] = wmma_f16(a, bb, acc[et]);
    }
  }

  const int hi8 = (lane >> 4) * 8;
#pragma unroll
  for (int et = 0; et < 4; ++et) {
    int e = et * 16 + (lane & 15);
#pragma unroll
    for (int r = 0; r < 8; ++r) {
      int rl = wid * 16 + hi8 + r;
      int s  = rows0 + rl;
      Vn[(size_t)(b * S_ + s) * D_ + h * HD_ + e] = (half_t)(acc[et][r] * Zs[rl]);
    }
  }
}

// ---------------------------------------------------------------------------
extern "C" void kernel_launch(void* const* d_in, const int* in_sizes, int n_in,
                              void* d_out, int out_size, void* d_ws, size_t ws_size,
                              hipStream_t stream) {
  const float* query = (const float*)d_in[0];
  const float* key   = (const float*)d_in[1];
  const float* value = (const float*)d_in[2];
  const unsigned char* mask = (const unsigned char*)d_in[3];
  const float* Wq = (const float*)d_in[4];
  const float* bq = (const float*)d_in[5];
  const float* Wk = (const float*)d_in[6];
  const float* bk = (const float*)d_in[7];
  const float* Wv = (const float*)d_in[8];
  const float* bv = (const float*)d_in[9];
  const float* Wo = (const float*)d_in[10];
  const float* bo = (const float*)d_in[11];

  const int M = B_ * S_;   // 16384
  const int N = D_;        // 1024
  const int K = D_;        // 1024

  char* ws = (char*)d_ws;
  half_t* qf  = (half_t*)ws;  ws += (size_t)M * D_ * sizeof(half_t);
  half_t* kf  = (half_t*)ws;  ws += (size_t)M * D_ * sizeof(half_t);
  half_t* vf  = (half_t*)ws;  ws += (size_t)M * D_ * sizeof(half_t);
  half_t* vn  = (half_t*)ws;  ws += (size_t)M * D_ * sizeof(half_t);
  half_t* wq6 = (half_t*)ws;  ws += (size_t)D_ * D_ * sizeof(half_t);
  half_t* wk6 = (half_t*)ws;  ws += (size_t)D_ * D_ * sizeof(half_t);
  half_t* wv6 = (half_t*)ws;  ws += (size_t)D_ * D_ * sizeof(half_t);
  half_t* wo6 = (half_t*)ws;  ws += (size_t)D_ * D_ * sizeof(half_t);
  float*  KVb = (float*)ws;   ws += (size_t)B_ * H_ * HD_ * HD_ * sizeof(float);
  float*  Ksm = (float*)ws;

  const int nW = D_ * D_;
  const int gcv = nW / (256 * 8);
  cvt_f32_f16<<<gcv, 256, 0, stream>>>(Wq, wq6, nW);
  cvt_f32_f16<<<gcv, 256, 0, stream>>>(Wk, wk6, nW);
  cvt_f32_f16<<<gcv, 256, 0, stream>>>(Wv, wv6, nW);
  cvt_f32_f16<<<gcv, 256, 0, stream>>>(Wo, wo6, nW);

  dim3 gproj(N / 128, M / 128);
  proj_gemm<float,  true,  false><<<gproj, 256, 0, stream>>>(query, wq6, bq, qf, M, N, K);
  proj_gemm<float,  true,  false><<<gproj, 256, 0, stream>>>(key,   wk6, bk, kf, M, N, K);
  proj_gemm<float,  false, false><<<gproj, 256, 0, stream>>>(value, wv6, bv, vf, M, N, K);

  kv_ksum_kernel<<<B_ * H_, 256, 0, stream>>>(kf, vf, mask, KVb, Ksm);

  apply_kv_kernel<<<dim3(S_ / 128, B_ * H_), 256, 0, stream>>>(qf, KVb, Ksm, vn);

  proj_gemm<half_t, false, true><<<gproj, 256, 0, stream>>>(vn, wo6, bo, d_out, M, N, K);
}
